// RelationalGATLayer_28484223107793
// MI455X (gfx1250) — matrix operand order repaired and checked by reading, not compiled, not run on hardware
//
#include <hip/hip_runtime.h>
#include <cstdint>
#include <cstddef>

typedef __attribute__((ext_vector_type(16))) __bf16  v16bf;
typedef __attribute__((ext_vector_type(8)))  float   v8f;
typedef __attribute__((ext_vector_type(4)))  float   f32x4;
typedef __attribute__((ext_vector_type(4)))  unsigned int u32x4;
typedef __attribute__((ext_vector_type(8)))  unsigned int u32x8;

#define N_NODES 30000
#define N_PAD   30080   /* padded to multiple of 128 rows (8 waves x 16) */
#define N_EDGES 300000
#define N_REL   8
#define N_HEAD  8
#define HEAD_D  32
#define IN_F    256
#define OUT_F   256     /* H*D */
#define NEG_SLOPE 0.2f
#define BPITCH  264     /* LDS row pitch in bf16 elems: 256 + 8 (TDM pad: 4 dwords / 128 dwords) */

// ---------------- utility kernels ----------------
__global__ void zero_f32_kernel(float* p, long n) {
  long i = (long)blockIdx.x * blockDim.x + threadIdx.x;
  if (i < n) p[i] = 0.0f;
}

__global__ void cvt_bf16_kernel(const float* __restrict__ in, __bf16* __restrict__ out, long n) {
  long i = (long)blockIdx.x * blockDim.x + threadIdx.x;
  if (i < n) out[i] = (__bf16)in[i];
}

// convert x with zero padding to N_PAD rows
__global__ void cvt_x_pad_kernel(const float* __restrict__ in, __bf16* __restrict__ out,
                                 long n_valid, long n_total) {
  long i = (long)blockIdx.x * blockDim.x + threadIdx.x;
  if (i < n_total) out[i] = (i < n_valid) ? (__bf16)in[i] : (__bf16)0.0f;
}

// u_src[r,h,f] = sum_d attn_src[h,d] * W[r, h*32+d, f]   (attention projected through W)
__global__ void build_u_kernel(const float* __restrict__ W,
                               const float* __restrict__ attn_src,
                               const float* __restrict__ attn_dst,
                               float* __restrict__ u_src,
                               float* __restrict__ u_dst) {
  int i = blockIdx.x * blockDim.x + threadIdx.x;          // over R*H*IN_F
  if (i >= N_REL * N_HEAD * IN_F) return;
  int f = i % IN_F;
  int t = i / IN_F;
  int h = t % N_HEAD;
  int r = t / N_HEAD;
  const float* wrow = W + ((size_t)r * OUT_F + h * HEAD_D) * IN_F + f;
  float ss = 0.0f, sd = 0.0f;
#pragma unroll 4
  for (int d = 0; d < HEAD_D; ++d) {
    float wv = wrow[(size_t)d * IN_F];
    ss += attn_src[h * HEAD_D + d] * wv;
    sd += attn_dst[h * HEAD_D + d] * wv;
  }
  u_src[i] = ss;
  u_dst[i] = sd;
}

// ---------------- WMMA projection GEMM with TDM-staged B ----------------
// hall[r, n, o] = sum_k xb[n,k] * wb[r, o, k]
// Block: 8 waves. One TDM tensor_load_to_lds stages the 64x256 bf16 B slab
// (columns col0..col0+63 of relation r) into LDS with a 264-elem padded pitch.
// Each wave computes a 16x64 tile; A (global) and B (LDS) are both software-
// pipelined one K-step ahead so WMMAs overlap the loads (no point-blank waits).
__global__ __launch_bounds__(256)
void wmma_proj_kernel(const __bf16* __restrict__ xb,
                      const __bf16* __restrict__ wb,
                      __bf16* __restrict__ hall) {
  __shared__ __bf16 Bs[64 * BPITCH];   // 33 KB of the 320 KB WGP LDS

  const int NT = OUT_F / 64;           // 4
  const int MG = N_PAD / 128;          // 235
  int bid = blockIdx.x;
  int nt  = bid % NT;
  int t   = bid / NT;
  int mg  = t % MG;
  int r   = t / MG;
  if (r >= N_REL) return;              // grid sized exactly; uniform guard

  const int col0 = nt * 64;
  const int lane = threadIdx.x & 31;
  const int wv   = threadIdx.x >> 5;   // wave id in block -> M tile
  const int row0 = mg * 128 + wv * 16;

  // ---- TDM: DMA the B slab into LDS (wave 0 only; TENSORcnt is per-wave) ----
  if (threadIdx.x < 32) {
    unsigned long long gaddr =
        (unsigned long long)(uintptr_t)(wb + ((size_t)r * OUT_F + col0) * IN_F);
    unsigned int laddr = (unsigned int)(uintptr_t)&Bs[0];

    // D# group 0: count=1 | lds_addr | global_addr[56:0] | type=2
    u32x4 g0;
    g0[0] = 1u;
    g0[1] = laddr;
    g0[2] = (unsigned int)gaddr;
    g0[3] = ((unsigned int)(gaddr >> 32) & 0x01ffffffu) | (2u << 30);

    // D# group 1: data_size=1(2B) | pad_enable, pad_interval=6 (128 dw), pad_amount=3 (4 dw)
    //             tensor_dim0=256, tensor_dim1=64, tile_dim0=256, tile_dim1=64, stride0=256
    u32x8 g1;
    g1[0] = (1u << 16) | (1u << 20) | (6u << 22) | (3u << 25);
    g1[1] = ((unsigned int)IN_F & 0xffffu) << 16;                 // tensor_dim0 lo
    g1[2] = (64u & 0xffffu) << 16;                                // dim0 hi=0 | tensor_dim1 lo
    g1[3] = ((unsigned int)IN_F & 0xffffu) << 16;                 // dim1 hi=0 | tile_dim0
    g1[4] = 64u;                                                  // tile_dim1 | tile_dim2=0
    g1[5] = (unsigned int)IN_F;                                   // stride0[31:0]
    g1[6] = 0u;                                                   // stride0 hi | stride1 lo
    g1[7] = 0u;                                                   // stride1 hi

    u32x4 g2 = (u32x4)(0u);  // dims 2/3 unused
    u32x4 g3 = (u32x4)(0u);

    asm volatile("tensor_load_to_lds %0, %1, %2, %3"
                 :: "s"(g0), "s"(g1), "s"(g2), "s"(g3)
                 : "memory");
    __builtin_amdgcn_s_wait_tensorcnt(0);
  }
  __syncthreads();

  // A fragment base: lane l<16 -> row row0+l, K {0..7,16..23}; l>=16 -> K {8..15,24..31}
  const __bf16* Abase = xb + (size_t)(row0 + (lane & 15)) * IN_F + ((lane >> 4) << 3);
  // B fragment from LDS: lane l<16 -> col-local j*16+l, K {0..7,16..23}; l>=16 -> K {8..15,24..31}
  const int bK = ((lane >> 4) << 3);
  const int bC = (lane & 15);
  const __bf16* Blds = &Bs[(size_t)bC * BPITCH + bK];

  v8f acc[4] = {{0.f,0.f,0.f,0.f,0.f,0.f,0.f,0.f},
                {0.f,0.f,0.f,0.f,0.f,0.f,0.f,0.f},
                {0.f,0.f,0.f,0.f,0.f,0.f,0.f,0.f},
                {0.f,0.f,0.f,0.f,0.f,0.f,0.f,0.f}};

  union AU { f32x4 q[2]; v16bf v; };
  union BU { f32x4 q[2]; v16bf v; };

  AU a_cur, a_nxt;
  BU b_cur[4], b_nxt[4];

  // prologue: load K-step 0
  a_cur.q[0] = *reinterpret_cast<const f32x4*>(Abase);
  a_cur.q[1] = *reinterpret_cast<const f32x4*>(Abase + 16);
#pragma unroll
  for (int j = 0; j < 4; ++j) {
    const __bf16* bp = Blds + (size_t)(j * 16) * BPITCH;
    b_cur[j].q[0] = *reinterpret_cast<const f32x4*>(bp);
    b_cur[j].q[1] = *reinterpret_cast<const f32x4*>(bp + 16);
  }

#pragma unroll
  for (int k0 = 0; k0 < IN_F; k0 += 32) {
    if (k0 + 32 < IN_F) {
      // issue next K-step loads before consuming current fragments
      a_nxt.q[0] = *reinterpret_cast<const f32x4*>(Abase + k0 + 32);
      a_nxt.q[1] = *reinterpret_cast<const f32x4*>(Abase + k0 + 48);
      __builtin_prefetch(Abase + k0 + 64, 0, 3);
#pragma unroll
      for (int j = 0; j < 4; ++j) {
        const __bf16* bp = Blds + (size_t)(j * 16) * BPITCH + (k0 + 32);
        b_nxt[j].q[0] = *reinterpret_cast<const f32x4*>(bp);
        b_nxt[j].q[1] = *reinterpret_cast<const f32x4*>(bp + 16);
      }
    }
#pragma unroll
    for (int j = 0; j < 4; ++j) {
      acc[j] = __builtin_amdgcn_wmma_f32_16x16x32_bf16(
          false, a_cur.v, false, b_cur[j].v, (short)0, acc[j], false, false);
    }
    a_cur = a_nxt;
#pragma unroll
    for (int j = 0; j < 4; ++j) b_cur[j] = b_nxt[j];
  }

  // C/D layout: VGPR v, lane<16 -> (M=v, N=lane); lane>=16 -> (M=v+8, N=lane-16)
  __bf16* outp = hall + (size_t)r * N_NODES * OUT_F;
  const int rb = (lane >> 4) << 3;
#pragma unroll
  for (int j = 0; j < 4; ++j) {
#pragma unroll
    for (int v = 0; v < 8; ++v) {
      int row = row0 + rb + v;
      if (row < N_NODES)
        outp[(size_t)row * OUT_F + (col0 + j * 16 + bC)] = (__bf16)acc[j][v];
    }
  }
}

// s_src[r,n,h] = x[n,:] . u_src[r,h,:]   (and s_dst likewise; shares x loads)
__global__ void node_scores_kernel(const float* __restrict__ x,
                                   const float* __restrict__ u_src,
                                   const float* __restrict__ u_dst,
                                   float* __restrict__ s_src,
                                   float* __restrict__ s_dst) {
  long i = (long)blockIdx.x * blockDim.x + threadIdx.x;   // n*64 + r*8 + h
  if (i >= (long)N_NODES * N_REL * N_HEAD) return;
  int  rh = (int)(i & 63);
  long n  = i >> 6;
  int  h  = rh & 7;
  int  r  = rh >> 3;
  const float* xp = x + n * IN_F;
  const float* us = u_src + (size_t)(r * N_HEAD + h) * IN_F;
  const float* ud = u_dst + (size_t)(r * N_HEAD + h) * IN_F;
  float ss = 0.0f, sd = 0.0f;
#pragma unroll 8
  for (int f = 0; f < IN_F; ++f) {
    float xv = xp[f];
    ss += xv * us[f];
    sd += xv * ud[f];
  }
  s_src[((size_t)r * N_NODES + n) * N_HEAD + h] = ss;
  s_dst[((size_t)r * N_NODES + n) * N_HEAD + h] = sd;
}

// leaky-relu logits + segment max (int-compare trick; amax zero-init == clamp at 0)
__global__ void edge_scores_kernel(const int* __restrict__ ei, const int* __restrict__ et,
                                   const float* __restrict__ s_src, const float* __restrict__ s_dst,
                                   float* __restrict__ wbuf, int* __restrict__ amax) {
  long i = (long)blockIdx.x * blockDim.x + threadIdx.x;   // e*8 + h
  if (i >= (long)N_EDGES * N_HEAD) return;
  int  h = (int)(i & 7);
  long e = i >> 3;
  int src = ei[e];
  int dst = ei[N_EDGES + e];
  int r   = et[e];
  float s = s_src[((size_t)r * N_NODES + src) * N_HEAD + h] +
            s_dst[((size_t)r * N_NODES + dst) * N_HEAD + h];
  s = s > 0.0f ? s : NEG_SLOPE * s;
  wbuf[i] = s;
  atomicMax(&amax[dst * N_HEAD + h], __float_as_int(s));
}

__global__ void edge_exp_kernel(const int* __restrict__ ei,
                                float* __restrict__ wbuf,
                                const int* __restrict__ amax,
                                float* __restrict__ denom) {
  long i = (long)blockIdx.x * blockDim.x + threadIdx.x;
  if (i >= (long)N_EDGES * N_HEAD) return;
  int  h = (int)(i & 7);
  long e = i >> 3;
  int dst = ei[N_EDGES + e];
  float m = __int_as_float(amax[dst * N_HEAD + h]);
  float w = __expf(wbuf[i] - m);
  wbuf[i] = w;
  atomicAdd(&denom[dst * N_HEAD + h], w);
}

// msg scatter: out[dst, hd] += (w/denom) * h_all_bf16[r, src, hd]
__global__ void edge_scatter_kernel(const int* __restrict__ ei, const int* __restrict__ et,
                                    const float* __restrict__ wbuf,
                                    const float* __restrict__ denom,
                                    const __bf16* __restrict__ hall,
                                    float* __restrict__ out) {
  long i = (long)blockIdx.x * blockDim.x + threadIdx.x;   // e*256 + hd
  if (i >= (long)N_EDGES * OUT_F) return;
  int  hd = (int)(i & 255);
  long e  = i >> 8;
  int h   = hd >> 5;
  int src = ei[e];
  int dst = ei[N_EDGES + e];
  int r   = et[e];
  float coef = wbuf[e * N_HEAD + h] / (denom[dst * N_HEAD + h] + 1e-12f);
  float v = (float)hall[((size_t)r * N_NODES + src) * OUT_F + hd];
  atomicAdd(&out[(size_t)dst * OUT_F + hd], coef * v);
}

// ---------------- host launch ----------------
extern "C" void kernel_launch(void* const* d_in, const int* in_sizes, int n_in,
                              void* d_out, int out_size, void* d_ws, size_t ws_size,
                              hipStream_t stream) {
  (void)in_sizes; (void)n_in; (void)out_size; (void)ws_size;
  const float* x        = (const float*)d_in[0];
  const float* W        = (const float*)d_in[1];
  const float* attn_src = (const float*)d_in[2];
  const float* attn_dst = (const float*)d_in[3];
  const int*   ei       = (const int*)d_in[4];
  const int*   et       = (const int*)d_in[5];
  float*       out      = (float*)d_out;

  // workspace carve (256B-aligned)
  char*  ws  = (char*)d_ws;
  size_t off = 0;
  auto carve = [&](size_t bytes) -> void* {
    off = (off + 255) & ~(size_t)255;
    void* p = ws + off;
    off += bytes;
    return p;
  };
  __bf16* xb    = (__bf16*)carve((size_t)N_PAD * IN_F * sizeof(__bf16));
  __bf16* wb    = (__bf16*)carve((size_t)N_REL * OUT_F * IN_F * sizeof(__bf16));
  float*  usrc  = (float*) carve((size_t)N_REL * N_HEAD * IN_F * sizeof(float));
  float*  udst  = (float*) carve((size_t)N_REL * N_HEAD * IN_F * sizeof(float));
  __bf16* hall  = (__bf16*)carve((size_t)N_REL * N_NODES * OUT_F * sizeof(__bf16)); // 123MB: L2-resident
  float*  ssrc  = (float*) carve((size_t)N_REL * N_NODES * N_HEAD * sizeof(float));
  float*  sdst  = (float*) carve((size_t)N_REL * N_NODES * N_HEAD * sizeof(float));
  int*    amax  = (int*)   carve((size_t)N_NODES * N_HEAD * sizeof(int));
  float*  denom = (float*) carve((size_t)N_NODES * N_HEAD * sizeof(float));
  float*  wbuf  = (float*) carve((size_t)N_EDGES * N_HEAD * sizeof(float));

  const int B = 256;
  auto blocks = [](long n, int b) { return (unsigned)((n + b - 1) / b); };

  // zero accumulator surfaces every call (harness poisons, does not re-poison)
  zero_f32_kernel<<<blocks((long)N_NODES * OUT_F, B), B, 0, stream>>>(out, (long)N_NODES * OUT_F);
  zero_f32_kernel<<<blocks((long)N_NODES * N_HEAD, B), B, 0, stream>>>((float*)amax, (long)N_NODES * N_HEAD);
  zero_f32_kernel<<<blocks((long)N_NODES * N_HEAD, B), B, 0, stream>>>(denom, (long)N_NODES * N_HEAD);

  // convert operands to bf16 (x zero-padded to N_PAD rows)
  cvt_x_pad_kernel<<<blocks((long)N_PAD * IN_F, B), B, 0, stream>>>(
      x, xb, (long)N_NODES * IN_F, (long)N_PAD * IN_F);
  cvt_bf16_kernel<<<blocks((long)N_REL * OUT_F * IN_F, B), B, 0, stream>>>(
      W, wb, (long)N_REL * OUT_F * IN_F);

  // attention vectors projected through W (kills the h_dst / h_src vector gathers for scores)
  build_u_kernel<<<blocks((long)N_REL * N_HEAD * IN_F, B), B, 0, stream>>>(W, attn_src, attn_dst, usrc, udst);

  // main WMMA projection: R * (OUT/64) * (N_PAD/128) blocks, 8 waves per block
  {
    unsigned nblocks = (unsigned)(N_REL * (OUT_F / 64) * (N_PAD / 128));   // 7520
    wmma_proj_kernel<<<nblocks, 256, 0, stream>>>(xb, wb, hall);
  }

  node_scores_kernel<<<blocks((long)N_NODES * N_REL * N_HEAD, B), B, 0, stream>>>(x, usrc, udst, ssrc, sdst);
  edge_scores_kernel<<<blocks((long)N_EDGES * N_HEAD, B), B, 0, stream>>>(ei, et, ssrc, sdst, wbuf, amax);
  edge_exp_kernel<<<blocks((long)N_EDGES * N_HEAD, B), B, 0, stream>>>(ei, wbuf, amax, denom);
  edge_scatter_kernel<<<blocks((long)N_EDGES * OUT_F, B), B, 0, stream>>>(ei, et, wbuf, denom, hall, out);
}
